// SpatialCrossAttnCSPLayer_86234353369158
// MI455X (gfx1250) — matrix operand order repaired and checked by reading, not compiled, not run on hardware
//
#include <hip/hip_runtime.h>
#include <hip/hip_bf16.h>

// ---------------------------------------------------------------------------
// Types / helpers
// ---------------------------------------------------------------------------
typedef __attribute__((ext_vector_type(16))) _Float16 v16h;
typedef __attribute__((ext_vector_type(8)))  float    v8f;
typedef __attribute__((ext_vector_type(4)))  int      v4i;

__device__ __forceinline__ float silu_f(float x) {
    return x / (1.0f + __expf(-x));
}

__device__ __forceinline__ v8f wmma_f16(v16h a, v16h b, v8f c) {
    // D = A(16x32 f16) * B(32x16 f16) + C(16x16 f32)
    return __builtin_amdgcn_wmma_f32_16x16x32_f16(
        /*neg_a=*/false, a, /*neg_b=*/false, b,
        /*c_mod=*/(short)0, c, /*reuse_a=*/false, /*reuse_b=*/false);
}

// Load a 16x32 A fragment from LDS, row-major f16 with row stride ldk halves.
// 16-bit A layout: lanes 0-15 hold M=lane; halves 0..7 -> K=0..7, 8..15 -> 16..23;
// lanes 16-31: halves 0..7 -> K=8..15, 8..15 -> 24..31.
// Per VGPR v (2 halves): kbase = 2v + 8g + (v>=4 ? 8 : 0)  (consecutive K pair).
__device__ __forceinline__ v16h lds_a_frag(const _Float16* sA, int ldk, int k0) {
    int lane = threadIdx.x & 31;
    int m = lane & 15, g = lane >> 4;
    union { v16h h; unsigned u[8]; } f;
    const _Float16* row = sA + m * ldk + k0;
#pragma unroll
    for (int v = 0; v < 8; ++v) {
        int kb = 2 * v + 8 * g + ((v >= 4) ? 8 : 0);
        f.u[v] = *(const unsigned*)(row + kb);   // ds_load_b32 of K,K+1 pair
    }
    return f.h;
}

// Load a 32x16 B fragment from fragment-packed weights in GLOBAL memory:
// packed[((kt*ntiles + nt)*32 + lane)*16 .. +15]  (32 contiguous bytes/lane).
__device__ __forceinline__ v16h gbl_b_frag(const _Float16* packed, int ntiles,
                                           int kt, int nt) {
    int lane = threadIdx.x & 31;
    const v16h* p =
        (const v16h*)(packed + (((size_t)kt * ntiles + nt) * 32 + lane) * 16);
    return *p;   // two global_load_b128
}

// Same fragment-packed layout, but staged in LDS.
__device__ __forceinline__ v16h lds_b_frag(const _Float16* sW, int ntiles,
                                           int kt, int nt) {
    int lane = threadIdx.x & 31;
    const v16h* p = (const v16h*)(sW + (((kt * ntiles + nt) * 32) + lane) * 16);
    return *p;   // two ds_load_b128
}

// Async global->LDS copy of 16 bytes per lane (gfx1250 ASYNCcnt-tracked path).
// Builtin signature (from probe): (AS1 int4* src, AS3 int4* lds, Imm off, Imm cpol).
// Per ISA 10.2, low 32 bits of a generic LDS pointer are the LDS byte offset.
__device__ __forceinline__ void async_g2l_b128(unsigned lds_off, const void* gsrc) {
#if __has_builtin(__builtin_amdgcn_global_load_async_to_lds_b128)
    __builtin_amdgcn_global_load_async_to_lds_b128(
        (__attribute__((address_space(1))) v4i*)(size_t)gsrc,
        (__attribute__((address_space(3))) v4i*)(size_t)lds_off, 0, 0);
#else
    unsigned long long ga = (unsigned long long)gsrc;
    asm volatile("global_load_async_to_lds_b128 %0, %1, off"
                 :: "v"(lds_off), "v"(ga)
                 : "memory");
#endif
}

__device__ __forceinline__ void wait_asynccnt0() {
#if __has_builtin(__builtin_amdgcn_s_wait_asynccnt)
    __builtin_amdgcn_s_wait_asynccnt(0);
#else
    asm volatile("s_wait_asynccnt 0x0" ::: "memory");
#endif
}

__device__ __forceinline__ void atomic_add_f32(float* p, float v) {
    unsafeAtomicAdd(p, v);   // native global_atomic_add_f32
}

// ---------------------------------------------------------------------------
// Kernel: zero scatter buffers
// ---------------------------------------------------------------------------
__global__ void k_zero(float* s, float* cnt, int n_s, int n_c) {
    int i = blockIdx.x * blockDim.x + threadIdx.x;
    int stride = gridDim.x * blockDim.x;
    for (int j = i; j < n_s; j += stride) s[j] = 0.0f;
    for (int j = i; j < n_c; j += stride) cnt[j] = 0.0f;
}

// ---------------------------------------------------------------------------
// Kernel: pack an f32 [Kdim, Ndim] weight into WMMA-B fragment layout (f16).
// B layout (32x16, 16-bit): half slot h, lane group g: K = h + 8*g + (h>=8 ? 8 : 0), N = lane%16.
// ---------------------------------------------------------------------------
__global__ void k_pack(const float* __restrict__ W, _Float16* __restrict__ out,
                       int Kdim, int Ndim) {
    int ktiles = (Kdim + 31) >> 5;
    int ntiles = (Ndim + 15) >> 4;
    int idx = blockIdx.x * blockDim.x + threadIdx.x;
    int total = ktiles * ntiles * 32;
    if (idx >= total) return;
    int lane = idx & 31;
    int frag = idx >> 5;
    int nt = frag % ntiles;
    int kt = frag / ntiles;
    int n = nt * 16 + (lane & 15);
    int g = lane >> 4;
    _Float16* dst = out + ((size_t)frag * 32 + lane) * 16;
#pragma unroll
    for (int h = 0; h < 16; ++h) {
        int k = kt * 32 + h + 8 * g + ((h >= 8) ? 8 : 0);
        float w = (k < Kdim && n < Ndim) ? W[(size_t)k * Ndim + n] : 0.0f;
        dst[h] = (_Float16)w;
    }
}

// ---------------------------------------------------------------------------
// Kernel: K/V projections:  Kmat = cond @ wk + bk,  Vmat = cond @ wv + bv
// ---------------------------------------------------------------------------
__global__ void k_kv(const float* __restrict__ cond,
                     const _Float16* __restrict__ pwk, const float* __restrict__ bk,
                     const _Float16* __restrict__ pwv, const float* __restrict__ bv,
                     float* __restrict__ Kmat, float* __restrict__ Vmat, int BT) {
    __shared__ __align__(32) _Float16 sX[16][128];
    int r0 = blockIdx.x * 16;
    int tid = threadIdx.x;
    for (int i = tid; i < 16 * 128; i += 256) {
        int r = i >> 7, c = i & 127;
        int rr = min(r0 + r, BT - 1);
        sX[r][c] = (_Float16)cond[(size_t)rr * 128 + c];
    }
    __syncthreads();
    int wave = tid >> 5, lane = tid & 31;
    int nt = wave;
    int col = nt * 16 + (lane & 15);
    int g = lane >> 4;
    v8f accK = {}; v8f accV = {};
    for (int kt = 0; kt < 4; ++kt) {
        v16h a = lds_a_frag(&sX[0][0], 128, kt * 32);
        accK = wmma_f16(a, gbl_b_frag(pwk, 8, kt, nt), accK);
        accV = wmma_f16(a, gbl_b_frag(pwv, 8, kt, nt), accV);
    }
    float bbk = bk[col], bbv = bv[col];
#pragma unroll
    for (int r = 0; r < 8; ++r) {
        int row = r0 + r + 8 * g;
        if (row < BT) {
            Kmat[(size_t)row * 128 + col] = accK[r] + bbk;
            Vmat[(size_t)row * 128 + col] = accV[r] + bbv;
        }
    }
}

// ---------------------------------------------------------------------------
// Kernel: fused cross-attention with spatial bias. 16 nodes / block (8 waves).
// ---------------------------------------------------------------------------
__global__ void k_attn(const float* __restrict__ nodef, const int* __restrict__ n2g,
                       const float* __restrict__ frac,
                       const _Float16* __restrict__ pwq, const float* __restrict__ bq,
                       const _Float16* __restrict__ psw1, const float* __restrict__ sb1,
                       const _Float16* __restrict__ psw2, const float* __restrict__ sb2,
                       const _Float16* __restrict__ pwo, const float* __restrict__ bo,
                       const float* __restrict__ Kmat, const float* __restrict__ Vmat,
                       float* __restrict__ hout, int N) {
    __shared__ __align__(32) _Float16 sH[16][128];
    __shared__ __align__(32) float    sQ[16][128];
    __shared__ __align__(32) _Float16 sEnc[16][32];
    __shared__ __align__(32) _Float16 sB1[16][128];
    __shared__ float sBias[16][8];
    __shared__ float sSc[16][8];
    __shared__ float sAttn[16][8];
    __shared__ __align__(32) _Float16 sCA[16][128];

    int n0 = blockIdx.x * 16;
    int tid = threadIdx.x;

    for (int i = tid; i < 16 * 128; i += 256) {
        int r = i >> 7, c = i & 127;
        int n = min(n0 + r, N - 1);
        sH[r][c] = (_Float16)nodef[(size_t)n * 128 + c];
    }
    for (int i = tid; i < 16 * 32; i += 256) {
        int r = i >> 5, c = i & 31;
        int n = min(n0 + r, N - 1);
        float v = 0.0f;
        if (c < 24) {
            int cc = c; bool is_cos = (cc >= 12); if (is_cos) cc -= 12;
            int axis = cc >> 2, f = cc & 3;
            float freq = (float)(1 << f) * 3.14159265358979323846f;
            float ph = frac[(size_t)n * 3 + axis] * freq;
            v = is_cos ? __cosf(ph) : __sinf(ph);
        }
        sEnc[r][c] = (_Float16)v;
    }
    __syncthreads();

    int wave = tid >> 5, lane = tid & 31;
    int nt = wave;
    int col = nt * 16 + (lane & 15);
    int g = lane >> 4;

    // Q = h @ wq + bq  -> sQ (f32)
    {
        v8f acc = {};
        for (int kt = 0; kt < 4; ++kt)
            acc = wmma_f16(lds_a_frag(&sH[0][0], 128, kt * 32),
                           gbl_b_frag(pwq, 8, kt, nt), acc);
        float b = bq[col];
#pragma unroll
        for (int r = 0; r < 8; ++r) sQ[r + 8 * g][col] = acc[r] + b;
    }
    // bias MLP layer 1: silu(enc @ sw1 + sb1) -> sB1 (f16). K = 24 padded to 32.
    {
        v8f acc = {};
        acc = wmma_f16(lds_a_frag(&sEnc[0][0], 32, 0), gbl_b_frag(psw1, 8, 0, nt), acc);
        float b = sb1[col];
#pragma unroll
        for (int r = 0; r < 8; ++r) sB1[r + 8 * g][col] = (_Float16)silu_f(acc[r] + b);
    }
    __syncthreads();

    // bias MLP layer 2: sB1 @ sw2 + sb2 -> sBias [16,8] (wave 0 only; N pad to 16)
    if (wave == 0) {
        v8f acc = {};
        for (int kt = 0; kt < 4; ++kt)
            acc = wmma_f16(lds_a_frag(&sB1[0][0], 128, kt * 32),
                           gbl_b_frag(psw2, 1, kt, 0), acc);
        int c = lane & 15;
        if (c < 8) {
            float b = sb2[c];
#pragma unroll
            for (int r = 0; r < 8; ++r) sBias[r + 8 * g][c] = acc[r] + b;
        }
    }
    __syncthreads();

    // scores: one thread per (node, token)
    if (tid < 128) {
        int n = tid >> 3, t = tid & 7;
        int gg = n2g[min(n0 + n, N - 1)];
        const float* kr = Kmat + ((size_t)gg * 8 + t) * 128;
        float s = 0.0f;
        for (int d = 0; d < 128; ++d) s += sQ[n][d] * kr[d];
        sSc[n][t] = s * 0.08838834764831845f + sBias[n][t];  // 1/sqrt(128)
    }
    __syncthreads();
    // softmax over T=8: one thread per node
    if (tid < 16) {
        float mx = -1e30f;
        for (int t = 0; t < 8; ++t) mx = fmaxf(mx, sSc[tid][t]);
        float e[8], sum = 0.0f;
        for (int t = 0; t < 8; ++t) { e[t] = __expf(sSc[tid][t] - mx); sum += e[t]; }
        float inv = 1.0f / sum;
        for (int t = 0; t < 8; ++t) sAttn[tid][t] = e[t] * inv;
    }
    __syncthreads();
    // ca_pre = attn @ V -> sCA (f16)
    for (int i = tid; i < 16 * 128; i += 256) {
        int n = i >> 7, d = i & 127;
        int gg = n2g[min(n0 + n, N - 1)];
        const float* vb = Vmat + ((size_t)gg * 8) * 128 + d;
        float s = 0.0f;
#pragma unroll
        for (int t = 0; t < 8; ++t) s += sAttn[n][t] * vb[t * 128];
        sCA[n][d] = (_Float16)s;
    }
    __syncthreads();
    // h = node_features + ca_pre @ wo + bo
    {
        v8f acc = {};
        for (int kt = 0; kt < 4; ++kt)
            acc = wmma_f16(lds_a_frag(&sCA[0][0], 128, kt * 32),
                           gbl_b_frag(pwo, 8, kt, nt), acc);
        float b = bo[col];
#pragma unroll
        for (int r = 0; r < 8; ++r) {
            int row = n0 + r + 8 * g;
            if (row < N)
                hout[(size_t)row * 128 + col] =
                    nodef[(size_t)row * 128 + col] + acc[r] + b;
        }
    }
}

// ---------------------------------------------------------------------------
// Kernel: per-source edge count (for the scatter mean)
// ---------------------------------------------------------------------------
__global__ void k_count(const int* __restrict__ e_src, float* __restrict__ cnt, int E) {
    int e = blockIdx.x * blockDim.x + threadIdx.x;
    if (e < E) atomic_add_f32(&cnt[e_src[e]], 1.0f);
}

// ---------------------------------------------------------------------------
// Kernel: edge MLP with fused gather + SiLU + scatter-add.
// 32 edges x 128 cols per block (8 waves; each wave does 2 M-tiles so every
// B fragment feeds two WMMAs). Layer-2 weights (32KB) are DMA'd into LDS with
// async global->LDS loads overlapped with gather staging + layer-1 WMMAs.
// ---------------------------------------------------------------------------
__global__ void k_edge(const float* __restrict__ h, const int* __restrict__ edges,
                       const int* __restrict__ e2g, const float* __restrict__ frac,
                       const float* __restrict__ lat,
                       const _Float16* __restrict__ pew1, const float* __restrict__ eb1,
                       const _Float16* __restrict__ pew2, const float* __restrict__ eb2,
                       float* __restrict__ ssum, int E) {
    __shared__ __align__(32) _Float16 sA[32][288];
    __shared__ __align__(32) _Float16 sAct[32][128];
    __shared__ __align__(32) _Float16 sW2[4 * 8 * 32 * 16];   // 32KB packed ew2
    int e0 = blockIdx.x * 32;
    int tid = threadIdx.x;
    const int* e_src = edges;       // edges[0]
    const int* e_dst = edges + E;   // edges[1]

    // kick off async DMA of layer-2 weights into LDS (ASYNCcnt-tracked)
    {
        unsigned dst0 = (unsigned)(size_t)&sW2[0];
        const char* src = (const char*)pew2;
#pragma unroll
        for (int j = 0; j < 8; ++j) {
            int idx = tid + j * 256;                 // 2048 x 16B = 32KB
            async_g2l_b128(dst0 + idx * 16, src + (size_t)idx * 16);
        }
    }

    // stage gathered edge inputs as f16 (overlaps with the async DMA above)
    for (int i = tid; i < 32 * 288; i += 256) {
        int r = i / 288, c = i % 288;
        int e = min(e0 + r, E - 1);
        float v = 0.0f;
        if (c < 128) {
            v = h[(size_t)e_src[e] * 128 + c];
        } else if (c < 256) {
            v = h[(size_t)e_dst[e] * 128 + (c - 128)];
        } else if (c < 265) {
            int idx = c - 256; int i3 = idx / 3, k3 = idx % 3;
            const float* L = lat + (size_t)e2g[e] * 9;
            v = L[i3 * 3 + 0] * L[k3 * 3 + 0] + L[i3 * 3 + 1] * L[k3 * 3 + 1] +
                L[i3 * 3 + 2] * L[k3 * 3 + 2];
        } else if (c < 268) {
            int d = c - 265;
            float fd = frac[(size_t)e_dst[e] * 3 + d] - frac[(size_t)e_src[e] * 3 + d];
            v = fd - floorf(fd);   // mod 1.0
        }
        sA[r][c] = (_Float16)v;
    }
    __syncthreads();

    int wave = tid >> 5, lane = tid & 31;
    int nt = wave, col = nt * 16 + (lane & 15), g = lane >> 4;

    // layer 1: K = 268 padded to 288 (9 k-tiles); one B load feeds 2 WMMAs
    v8f acc0 = {}, acc1 = {};
    for (int kt = 0; kt < 9; ++kt) {
        v16h b = gbl_b_frag(pew1, 8, kt, nt);
        acc0 = wmma_f16(lds_a_frag(&sA[0][0],  288, kt * 32), b, acc0);
        acc1 = wmma_f16(lds_a_frag(&sA[16][0], 288, kt * 32), b, acc1);
    }
    float b1 = eb1[col];
#pragma unroll
    for (int r = 0; r < 8; ++r) {
        sAct[r + 8 * g][col]      = (_Float16)silu_f(acc0[r] + b1);
        sAct[16 + r + 8 * g][col] = (_Float16)silu_f(acc1[r] + b1);
    }
    wait_asynccnt0();          // layer-2 weights resident in LDS
    __syncthreads();

    // layer 2: B fragments now come from LDS
    v8f d0 = {}, d1 = {};
    for (int kt = 0; kt < 4; ++kt) {
        v16h b = lds_b_frag(sW2, 8, kt, nt);
        d0 = wmma_f16(lds_a_frag(&sAct[0][0],  128, kt * 32), b, d0);
        d1 = wmma_f16(lds_a_frag(&sAct[16][0], 128, kt * 32), b, d1);
    }
    float b2 = eb2[col];
#pragma unroll
    for (int r = 0; r < 8; ++r) {
        int m = r + 8 * g;
        int e = e0 + m;
        if (e < E)
            atomic_add_f32(&ssum[(size_t)e_src[e] * 128 + col], silu_f(d0[r] + b2));
        int e2 = e0 + 16 + m;
        if (e2 < E)
            atomic_add_f32(&ssum[(size_t)e_src[e2] * 128 + col], silu_f(d1[r] + b2));
    }
}

// ---------------------------------------------------------------------------
// Kernel: node MLP; out = node_features + silu(silu([h,agg]@nw1+nb1)@nw2+nb2)
// 32 nodes per block; each B fragment feeds two WMMAs.
// ---------------------------------------------------------------------------
__global__ void k_node(const float* __restrict__ h, const float* __restrict__ ssum,
                       const float* __restrict__ cnt, const float* __restrict__ nodef,
                       const _Float16* __restrict__ pnw1, const float* __restrict__ nb1,
                       const _Float16* __restrict__ pnw2, const float* __restrict__ nb2,
                       float* __restrict__ out, int N) {
    __shared__ __align__(32) _Float16 sA[32][256];
    __shared__ __align__(32) _Float16 sAct[32][128];
    int n0 = blockIdx.x * 32;
    int tid = threadIdx.x;
    for (int i = tid; i < 32 * 256; i += 256) {
        int r = i >> 8, c = i & 255;
        int n = min(n0 + r, N - 1);
        float v;
        if (c < 128) v = h[(size_t)n * 128 + c];
        else v = ssum[(size_t)n * 128 + (c - 128)] / fmaxf(cnt[n], 1.0f);
        sA[r][c] = (_Float16)v;
    }
    __syncthreads();

    int wave = tid >> 5, lane = tid & 31;
    int nt = wave, col = nt * 16 + (lane & 15), g = lane >> 4;

    v8f acc0 = {}, acc1 = {};
    for (int kt = 0; kt < 8; ++kt) {
        v16h b = gbl_b_frag(pnw1, 8, kt, nt);
        acc0 = wmma_f16(lds_a_frag(&sA[0][0],  256, kt * 32), b, acc0);
        acc1 = wmma_f16(lds_a_frag(&sA[16][0], 256, kt * 32), b, acc1);
    }
    float b1 = nb1[col];
#pragma unroll
    for (int r = 0; r < 8; ++r) {
        sAct[r + 8 * g][col]      = (_Float16)silu_f(acc0[r] + b1);
        sAct[16 + r + 8 * g][col] = (_Float16)silu_f(acc1[r] + b1);
    }
    __syncthreads();

    v8f d0 = {}, d1 = {};
    for (int kt = 0; kt < 4; ++kt) {
        v16h b = gbl_b_frag(pnw2, 8, kt, nt);
        d0 = wmma_f16(lds_a_frag(&sAct[0][0],  128, kt * 32), b, d0);
        d1 = wmma_f16(lds_a_frag(&sAct[16][0], 128, kt * 32), b, d1);
    }
    float b2 = nb2[col];
#pragma unroll
    for (int r = 0; r < 8; ++r) {
        int row = n0 + r + 8 * g;
        if (row < N)
            out[(size_t)row * 128 + col] =
                nodef[(size_t)row * 128 + col] + silu_f(d0[r] + b2);
        int row2 = n0 + 16 + r + 8 * g;
        if (row2 < N)
            out[(size_t)row2 * 128 + col] =
                nodef[(size_t)row2 * 128 + col] + silu_f(d1[r] + b2);
    }
}

// ---------------------------------------------------------------------------
// Host launcher
// ---------------------------------------------------------------------------
extern "C" void kernel_launch(void* const* d_in, const int* in_sizes, int n_in,
                              void* d_out, int out_size, void* d_ws, size_t ws_size,
                              hipStream_t stream) {
    const float* nodef = (const float*)d_in[0];
    const float* cond  = (const float*)d_in[1];
    const int*   n2g   = (const int*)d_in[2];
    const float* frac  = (const float*)d_in[3];
    const float* lat   = (const float*)d_in[4];
    const int*   edges = (const int*)d_in[5];
    const int*   e2g   = (const int*)d_in[6];
    const float* wq  = (const float*)d_in[7];   const float* bq  = (const float*)d_in[8];
    const float* wk  = (const float*)d_in[9];   const float* bk  = (const float*)d_in[10];
    const float* wv  = (const float*)d_in[11];  const float* bv  = (const float*)d_in[12];
    const float* wo  = (const float*)d_in[13];  const float* bo  = (const float*)d_in[14];
    const float* sw1 = (const float*)d_in[15];  const float* sb1 = (const float*)d_in[16];
    const float* sw2 = (const float*)d_in[17];  const float* sb2 = (const float*)d_in[18];
    const float* ew1 = (const float*)d_in[19];  const float* eb1 = (const float*)d_in[20];
    const float* ew2 = (const float*)d_in[21];  const float* eb2 = (const float*)d_in[22];
    const float* nw1 = (const float*)d_in[23];  const float* nb1 = (const float*)d_in[24];
    const float* nw2 = (const float*)d_in[25];  const float* nb2 = (const float*)d_in[26];

    const int N  = in_sizes[0] / 128;
    const int BT = in_sizes[1] / 128;     // B * T
    const int E  = in_sizes[6];

    // workspace carve-out (256B aligned slices)
    char* ws = (char*)d_ws;
    size_t off = 0;
    auto take = [&](size_t bytes) -> char* {
        char* p = ws + off;
        off = (off + bytes + 255) & ~(size_t)255;
        return p;
    };
    float* h_ws  = (float*)take((size_t)N * 128 * 4);
    float* s_ws  = (float*)take((size_t)N * 128 * 4);
    float* c_ws  = (float*)take((size_t)N * 4);
    float* Kmat  = (float*)take((size_t)BT * 128 * 4);
    float* Vmat  = (float*)take((size_t)BT * 128 * 4);

    auto frag_bytes = [](int Kdim, int Ndim) -> size_t {
        return (size_t)((Kdim + 31) / 32) * ((Ndim + 15) / 16) * 32 * 16 * 2;
    };
    _Float16* pwq  = (_Float16*)take(frag_bytes(128, 128));
    _Float16* pwk  = (_Float16*)take(frag_bytes(128, 128));
    _Float16* pwv  = (_Float16*)take(frag_bytes(128, 128));
    _Float16* pwo  = (_Float16*)take(frag_bytes(128, 128));
    _Float16* psw1 = (_Float16*)take(frag_bytes(24, 128));
    _Float16* psw2 = (_Float16*)take(frag_bytes(128, 8));
    _Float16* pew1 = (_Float16*)take(frag_bytes(268, 128));
    _Float16* pew2 = (_Float16*)take(frag_bytes(128, 128));
    _Float16* pnw1 = (_Float16*)take(frag_bytes(256, 128));
    _Float16* pnw2 = (_Float16*)take(frag_bytes(128, 128));

    // 1) zero scatter buffers
    k_zero<<<1024, 256, 0, stream>>>(s_ws, c_ws, N * 128, N);

    // 2) pack weights into WMMA-B fragment layout (f16)
    auto pack = [&](const float* W, _Float16* dst, int Kdim, int Ndim) {
        int total = ((Kdim + 31) / 32) * ((Ndim + 15) / 16) * 32;
        k_pack<<<(total + 255) / 256, 256, 0, stream>>>(W, dst, Kdim, Ndim);
    };
    pack(wq, pwq, 128, 128);
    pack(wk, pwk, 128, 128);
    pack(wv, pwv, 128, 128);
    pack(wo, pwo, 128, 128);
    pack(sw1, psw1, 24, 128);
    pack(sw2, psw2, 128, 8);
    pack(ew1, pew1, 268, 128);
    pack(ew2, pew2, 128, 128);
    pack(nw1, pnw1, 256, 128);
    pack(nw2, pnw2, 128, 128);

    // 3) K/V projections over cond tokens
    k_kv<<<(BT + 15) / 16, 256, 0, stream>>>(cond, pwk, bk, pwv, bv, Kmat, Vmat, BT);

    // 4) fused cross-attention -> h_ws
    k_attn<<<(N + 15) / 16, 256, 0, stream>>>(nodef, n2g, frac,
                                              pwq, bq, psw1, sb1, psw2, sb2,
                                              pwo, bo, Kmat, Vmat, h_ws, N);

    // 5) edge counts for scatter mean
    k_count<<<(E + 255) / 256, 256, 0, stream>>>(edges, c_ws, E);

    // 6) edge MLP with fused gather + scatter-add (async weight DMA inside)
    k_edge<<<(E + 31) / 32, 256, 0, stream>>>(h_ws, edges, e2g, frac, lat,
                                              pew1, eb1, pew2, eb2, s_ws, E);

    // 7) node MLP -> d_out
    k_node<<<(N + 31) / 32, 256, 0, stream>>>(h_ws, s_ws, c_ws, nodef,
                                              pnw1, nb1, pnw2, nb2,
                                              (float*)d_out, N);
}
